// EncoderBlock_14688788152934
// MI455X (gfx1250) — matrix operand order repaired and checked by reading, not compiled, not run on hardware
//
#include <hip/hip_runtime.h>

// ---------------- problem constants ----------------
#define B_   16
#define S_   1024
#define D_   512
#define H_   8
#define F_   2048
#define DH_  64
#define BS_  (B_ * S_)
#define EPS_ 1e-5f

// ---------------- vector types ----------------
typedef __attribute__((ext_vector_type(16))) __bf16 v16bf;
typedef __attribute__((ext_vector_type(8)))  __bf16 v8bf;
typedef __attribute__((ext_vector_type(8)))  float  v8f;

static __device__ __forceinline__ v16bf cat8(v8bf lo, v8bf hi) {
    union { v16bf v; v8bf h[2]; } u;
    u.h[0] = lo; u.h[1] = hi;
    return u.v;
}
static __device__ __forceinline__ v8f zero8() {
    v8f z;
#pragma unroll
    for (int i = 0; i < 8; ++i) z[i] = 0.f;
    return z;
}
static __device__ __forceinline__ v8f wmma_bf16(v16bf a, v16bf b, v8f c) {
    // D(f32 16x16) = A(bf16 16x32) * B(bf16 32x16) + C
    return __builtin_amdgcn_wmma_f32_16x16x32_bf16(
        /*neg_a=*/false, a, /*neg_b=*/false, b,
        /*c_mod=*/(short)0, c, /*reuse_a=*/false, /*reuse_b=*/false);
}

// ---- CDNA5 async global->LDS (ASYNCcnt-tracked DMA, bypasses VGPRs) --------
static __device__ __forceinline__ void async_g2lds_b128(const void* gptr, void* lptr) {
    // generic LDS pointer: low 32 bits are the LDS byte offset
    unsigned lds = (unsigned)(unsigned long long)lptr;
    asm volatile("global_load_async_to_lds_b128 %0, %1, off"
                 :: "v"(lds), "v"(gptr) : "memory");
}
static __device__ __forceinline__ void wait_async0() {
    asm volatile("s_wait_asynccnt 0x0" ::: "memory");
}

// ---------------- f32 -> bf16 convert (row-major copy) ----------------
__global__ void cvt_f32_bf16(const float* __restrict__ in, __bf16* __restrict__ out, size_t n) {
    size_t i = (size_t)blockIdx.x * blockDim.x + threadIdx.x;
    size_t stride = (size_t)gridDim.x * blockDim.x;
    for (; i < n; i += stride) out[i] = (__bf16)in[i];
}

// ---------------- fused convert + transpose: W[K][N] f32 -> WT[N][K] bf16 ----
__global__ __launch_bounds__(256)
void cvtT_f32_bf16(const float* __restrict__ in, __bf16* __restrict__ out, int K, int N) {
    __shared__ float tile[32][33];
    const int k0 = blockIdx.y * 32, n0 = blockIdx.x * 32;
    const int tx = threadIdx.x & 31, ty = threadIdx.x >> 5;  // 32 x 8
#pragma unroll
    for (int i = 0; i < 32; i += 8)
        tile[ty + i][tx] = in[(size_t)(k0 + ty + i) * N + n0 + tx];
    __syncthreads();
#pragma unroll
    for (int i = 0; i < 32; i += 8)
        out[(size_t)(n0 + ty + i) * K + k0 + tx] = (__bf16)tile[tx][ty + i];
}

// ---------------- V transpose: vb[b][s][h*64+dh] -> vt[(b*H+h)][dh][S] -------
__global__ __launch_bounds__(256)
void transpose_v(const __bf16* __restrict__ vsrc, __bf16* __restrict__ vt) {
    __shared__ __bf16 tile[32][33];
    const int bh = blockIdx.z, b = bh >> 3, h = bh & 7;
    const int s0 = blockIdx.x * 32, d0 = blockIdx.y * 32;   // d0 in {0,32}
    const int tx = threadIdx.x & 31, ty = threadIdx.x >> 5;
    const __bf16* src = vsrc + (size_t)b * S_ * D_ + (size_t)h * DH_;
#pragma unroll
    for (int i = 0; i < 32; i += 8)
        tile[ty + i][tx] = src[(size_t)(s0 + ty + i) * D_ + d0 + tx];
    __syncthreads();
    __bf16* dst = vt + (size_t)bh * DH_ * S_;
#pragma unroll
    for (int i = 0; i < 32; i += 8)
        dst[(size_t)(d0 + ty + i) * S_ + s0 + tx] = tile[tx][ty + i];
}

// ---------------- tiled bf16 WMMA GEMM ----------------
// C[M,N] = A[M,K] * WT[N,K]^T (+bias) (+relu); A, WT bf16 row-major.
// Block: 256 threads (8 waves). Block tile 128(M) x 128(N).
// Wave tile 64(M) x 32(N) = 4x2 WMMA tiles.
// Double-buffered LDS filled by async global->LDS DMA; 1 sync/K-step.
template <bool BIAS, bool RELU, bool OUTBF>
__global__ __launch_bounds__(256)
void gemm_bf16(const __bf16* __restrict__ A, const __bf16* __restrict__ WT,
               const float* __restrict__ bias,
               float* __restrict__ Cf, __bf16* __restrict__ Cb,
               int M, int N, int K) {
    __shared__ __align__(128) __bf16 sA[2][128 * 32];   // [m][k]
    __shared__ __align__(128) __bf16 sB[2][128 * 32];   // [n][k]

    const int t    = threadIdx.x;
    const int lane = t & 31;
    const int w    = t >> 5;
    const int ln   = lane & 15;
    const int hi   = lane >> 4;
    const int m0   = blockIdx.y * 128;
    const int n0   = blockIdx.x * 128;
    const int wm   = w >> 2;      // 0..1 -> 64-row strip
    const int wn   = w & 3;       // 0..3 -> 32-col strip

    v8f acc[4][2];
#pragma unroll
    for (int i = 0; i < 4; ++i)
#pragma unroll
        for (int j = 0; j < 2; ++j) acc[i][j] = zero8();

    // staging: 256 threads, 2 per row, 32B each (both tiles are [128][32] copies)
    const int srow = t >> 1, shalf = t & 1;
    const __bf16* aSrc = A  + (size_t)(m0 + srow) * K + shalf * 16;
    const __bf16* bSrc = WT + (size_t)(n0 + srow) * K + shalf * 16;
    const int sOff = srow * 32 + shalf * 16;

    // prologue: async-stage K-slab 0 into buffer 0
    async_g2lds_b128(aSrc,     sA[0] + sOff);
    async_g2lds_b128(aSrc + 8, sA[0] + sOff + 8);
    async_g2lds_b128(bSrc,     sB[0] + sOff);
    async_g2lds_b128(bSrc + 8, sB[0] + sOff + 8);
    wait_async0();
    __syncthreads();

    const int nk = K / 32;
    for (int ki = 0; ki < nk; ++ki) {
        const int cur = ki & 1;
        if (ki + 1 < nk) {   // async-prefetch next K-slab into the other buffer
            const __bf16* an = aSrc + (size_t)(ki + 1) * 32;
            const __bf16* bn = bSrc + (size_t)(ki + 1) * 32;
            async_g2lds_b128(an,     sA[cur ^ 1] + sOff);
            async_g2lds_b128(an + 8, sA[cur ^ 1] + sOff + 8);
            async_g2lds_b128(bn,     sB[cur ^ 1] + sOff);
            async_g2lds_b128(bn + 8, sB[cur ^ 1] + sOff + 8);
        }
        v16bf af[4], bfm[2];
#pragma unroll
        for (int tm = 0; tm < 4; ++tm) {
            const __bf16* p = sA[cur] + (wm * 64 + tm * 16 + ln) * 32;
            af[tm] = *(const v16bf*)(p + 16 * hi);
        }
#pragma unroll
        for (int tn = 0; tn < 2; ++tn)
            bfm[tn] = *(const v16bf*)(sB[cur] + (wn * 32 + tn * 16 + ln) * 32 + 16 * hi);
#pragma unroll
        for (int tm = 0; tm < 4; ++tm)
#pragma unroll
            for (int tn = 0; tn < 2; ++tn)
                acc[tm][tn] = wmma_bf16(af[tm], bfm[tn], acc[tm][tn]);
        wait_async0();
        __syncthreads();
    }

    // epilogue
#pragma unroll
    for (int tm = 0; tm < 4; ++tm) {
#pragma unroll
        for (int tn = 0; tn < 2; ++tn) {
            const int n  = n0 + wn * 32 + tn * 16 + ln;
            const int mb = m0 + wm * 64 + tm * 16 + 8 * hi;
            float bv = BIAS ? bias[n] : 0.f;
#pragma unroll
            for (int r = 0; r < 8; ++r) {
                float v = acc[tm][tn][r] + bv;
                if (RELU) v = fmaxf(v, 0.f);
                size_t idx = (size_t)(mb + r) * N + n;
                if (OUTBF) Cb[idx] = (__bf16)v;
                else       Cf[idx] = v;
            }
        }
    }
}

// ---------------- flash attention ----------------
// q,k: bf16 [B,S,D] heads packed; vt: bf16 [B*H][DH][S]; ctx: bf16 [B,S,D].
// One wave per 16-row Q tile per (b,h); 4 waves per block; 64 keys/iteration.
__global__ __launch_bounds__(128)
void attn_kernel(const __bf16* __restrict__ q, const __bf16* __restrict__ k,
                 const __bf16* __restrict__ vt, const int* __restrict__ valid_lens,
                 __bf16* __restrict__ ctx) {
    __shared__ __align__(128) __bf16 sP[4][16 * 64];    // per-wave P: [row][key_local]

    const int w    = threadIdx.x >> 5;
    const int lane = threadIdx.x & 31;
    const int ln   = lane & 15;
    const int hi   = lane >> 4;
    const int bh   = blockIdx.y;           // 0..B*H-1
    const int b    = bh >> 3;              // H_ == 8
    const int h    = bh & 7;
    const int s0   = (blockIdx.x * 4 + w) * 16;

    const size_t base  = (size_t)b * S_ * D_ + (size_t)h * DH_;
    const __bf16* vtb  = vt + (size_t)bh * DH_ * S_;

    // Q fragments (A layout), registers for the whole pass
    v16bf qa[2];
#pragma unroll
    for (int st = 0; st < 2; ++st) {
        const __bf16* qp = q + base + (size_t)(s0 + ln) * D_ + st * 32;
        qa[st] = *(const v16bf*)(qp + 16 * hi);
    }

    const int valid = valid_lens[b];
    const int kend  = min(S_, (valid + 63) & ~63);

    float m8[8], l8[8];
    v8f o[4];
#pragma unroll
    for (int r = 0; r < 8; ++r) { m8[r] = -1e30f; l8[r] = 0.f; }
#pragma unroll
    for (int tn = 0; tn < 4; ++tn) o[tn] = zero8();

    for (int kb = 0; kb < kend; kb += 64) {
        // scores: 16 q-rows x 64 keys = 4 tiles, K-dim = dh (2 steps of 32)
        v8f sc[4];
#pragma unroll
        for (int tk = 0; tk < 4; ++tk) {
            sc[tk] = zero8();
#pragma unroll
            for (int st = 0; st < 2; ++st) {
                const __bf16* kp = k + base + (size_t)(kb + tk * 16 + ln) * D_ + st * 32 + 16 * hi;
                sc[tk] = wmma_bf16(qa[st], *(const v16bf*)kp, sc[tk]);
            }
            const int key = kb + tk * 16 + ln;       // this lane's key column
            const bool masked = key >= valid;
#pragma unroll
            for (int r = 0; r < 8; ++r) {
                float s = sc[tk][r] * 0.125f;        // 1/sqrt(64)
                sc[tk][r] = masked ? -1e9f : s;
            }
        }

        // online softmax (row = r + 8*hi; reduce across 16-lane half)
#pragma unroll
        for (int r = 0; r < 8; ++r) {
            float tmax = fmaxf(fmaxf(sc[0][r], sc[1][r]), fmaxf(sc[2][r], sc[3][r]));
#pragma unroll
            for (int mk = 8; mk >= 1; mk >>= 1) tmax = fmaxf(tmax, __shfl_xor(tmax, mk, 32));
            const float mn = fmaxf(m8[r], tmax);
            const float al = __expf(m8[r] - mn);
            m8[r] = mn;
            float p[4];
            float rs = 0.f;
#pragma unroll
            for (int tk = 0; tk < 4; ++tk) { p[tk] = __expf(sc[tk][r] - mn); rs += p[tk]; }
#pragma unroll
            for (int mk = 8; mk >= 1; mk >>= 1) rs += __shfl_xor(rs, mk, 32);
            l8[r] = l8[r] * al + rs;
            const int row = r + 8 * hi;
#pragma unroll
            for (int tk = 0; tk < 4; ++tk)
                sP[w][row * 64 + tk * 16 + ln] = (__bf16)p[tk];
#pragma unroll
            for (int tn = 0; tn < 4; ++tn) o[tn][r] *= al;
        }

        // ctx += P(16x64) * V(64x64): P A-frags via LDS, V^T B-frags from global
#pragma unroll
        for (int j = 0; j < 2; ++j) {            // 32-key sub-blocks
            const __bf16* pp = &sP[w][ln * 64 + j * 32];
            v16bf pf = *(const v16bf*)(pp + 16 * hi);
#pragma unroll
            for (int tn = 0; tn < 4; ++tn) {
                const __bf16* bp = vtb + (size_t)(tn * 16 + ln) * S_ + kb + j * 32 + 16 * hi;
                o[tn] = wmma_bf16(pf, *(const v16bf*)bp, o[tn]);
            }
        }
    }

    // normalize and store ctx (heads re-merged into D)
#pragma unroll
    for (int tn = 0; tn < 4; ++tn) {
#pragma unroll
        for (int r = 0; r < 8; ++r) {
            const int row = r + 8 * hi;
            float val = o[tn][r] / l8[r];
            ctx[base + (size_t)(s0 + row) * D_ + tn * 16 + ln] = (__bf16)val;
        }
    }
}

// ---------------- fused add + LayerNorm ----------------
template <bool WRITE_BF>
__global__ __launch_bounds__(128)
void addnorm_kernel(const float* __restrict__ x, const float* __restrict__ a,
                    const float* __restrict__ g, const float* __restrict__ bta,
                    float* __restrict__ y, __bf16* __restrict__ yb) {
    const int row  = blockIdx.x;
    const int t    = threadIdx.x;
    const int lane = t & 31;
    const int w    = t >> 5;
    const float* xr = x + (size_t)row * D_;
    const float* ar = a + (size_t)row * D_;

    float vals[4];
    float sum = 0.f, sq = 0.f;
#pragma unroll
    for (int i = 0; i < 4; ++i) {
        const int c = t + i * 128;
        const float s = xr[c] + ar[c];
        vals[i] = s;
        sum += s;
        sq  += s * s;
    }
#pragma unroll
    for (int mk = 16; mk >= 1; mk >>= 1) {
        sum += __shfl_xor(sum, mk, 32);
        sq  += __shfl_xor(sq, mk, 32);
    }
    __shared__ float red[8];
    if (lane == 0) { red[w * 2] = sum; red[w * 2 + 1] = sq; }
    __syncthreads();
    float ts = 0.f, tq = 0.f;
#pragma unroll
    for (int i = 0; i < 4; ++i) { ts += red[i * 2]; tq += red[i * 2 + 1]; }
    const float mu  = ts * (1.f / D_);
    const float var = tq * (1.f / D_) - mu * mu;
    const float inv = rsqrtf(var + EPS_);
#pragma unroll
    for (int i = 0; i < 4; ++i) {
        const int c = t + i * 128;
        const float nv = (vals[i] - mu) * inv * g[c] + bta[c];
        y[(size_t)row * D_ + c] = nv;
        if (WRITE_BF) yb[(size_t)row * D_ + c] = (__bf16)nv;
    }
}

// ---------------- host side ----------------
extern "C" void kernel_launch(void* const* d_in, const int* in_sizes, int n_in,
                              void* d_out, int out_size, void* d_ws, size_t ws_size,
                              hipStream_t stream) {
    const float* x   = (const float*)d_in[0];
    const int*   vl  = (const int*)d_in[1];
    const float* wq  = (const float*)d_in[2];
    const float* wk  = (const float*)d_in[3];
    const float* wv  = (const float*)d_in[4];
    const float* wo  = (const float*)d_in[5];
    const float* w1  = (const float*)d_in[6];
    const float* b1  = (const float*)d_in[7];
    const float* w2  = (const float*)d_in[8];
    const float* b2  = (const float*)d_in[9];
    const float* g1  = (const float*)d_in[10];
    const float* be1 = (const float*)d_in[11];
    const float* g2  = (const float*)d_in[12];
    const float* be2 = (const float*)d_in[13];
    float* out = (float*)d_out;

    char*  ws  = (char*)d_ws;
    size_t off = 0;
    auto alloc = [&](size_t bytes) -> char* {
        char* p = ws + off;
        off += (bytes + 255) & ~(size_t)255;
        return p;
    };
    __bf16* xb   = (__bf16*)alloc((size_t)BS_ * D_ * 2);
    __bf16* qb   = (__bf16*)alloc((size_t)BS_ * D_ * 2);
    __bf16* kb   = (__bf16*)alloc((size_t)BS_ * D_ * 2);
    __bf16* vb   = (__bf16*)alloc((size_t)BS_ * D_ * 2);
    __bf16* hb   = xb;  // FFN hidden (BS x F bf16 == 64 MB) aliases [xb..vb], free by then
    __bf16* ctxb = (__bf16*)alloc((size_t)BS_ * D_ * 2);
    __bf16* vt   = (__bf16*)alloc((size_t)BS_ * D_ * 2);      // V^T [B*H][DH][S]
    __bf16* wqT  = (__bf16*)alloc((size_t)D_ * D_ * 2);
    __bf16* wkT  = (__bf16*)alloc((size_t)D_ * D_ * 2);
    __bf16* wvT  = (__bf16*)alloc((size_t)D_ * D_ * 2);
    __bf16* woT  = (__bf16*)alloc((size_t)D_ * D_ * 2);
    __bf16* w1T  = (__bf16*)alloc((size_t)D_ * F_ * 2);
    __bf16* w2T  = (__bf16*)alloc((size_t)F_ * D_ * 2);
    float*  tf   = (float*)alloc((size_t)BS_ * D_ * 4);  // attn-out, later FFN out
    float*  y    = (float*)alloc((size_t)BS_ * D_ * 4);
    __bf16* yb   = (__bf16*)alloc((size_t)BS_ * D_ * 2);

    // activations: plain convert
    {
        size_t n = (size_t)BS_ * D_;
        cvt_f32_bf16<<<2048, 256, 0, stream>>>(x, xb, n);
    }
    // weights: fused convert + transpose  ([K][N] f32 -> [N][K] bf16)
    {
        dim3 gDD(D_ / 32, D_ / 32);
        cvtT_f32_bf16<<<gDD, 256, 0, stream>>>(wq, wqT, D_, D_);
        cvtT_f32_bf16<<<gDD, 256, 0, stream>>>(wk, wkT, D_, D_);
        cvtT_f32_bf16<<<gDD, 256, 0, stream>>>(wv, wvT, D_, D_);
        cvtT_f32_bf16<<<gDD, 256, 0, stream>>>(wo, woT, D_, D_);
        dim3 gDF(F_ / 32, D_ / 32);
        cvtT_f32_bf16<<<gDF, 256, 0, stream>>>(w1, w1T, D_, F_);
        dim3 gFD(D_ / 32, F_ / 32);
        cvtT_f32_bf16<<<gFD, 256, 0, stream>>>(w2, w2T, F_, D_);
    }

    // QKV projections: [BS,D] x [D,D] -> bf16
    {
        dim3 grid(D_ / 128, BS_ / 128);
        gemm_bf16<false, false, true><<<grid, 256, 0, stream>>>(xb, wqT, nullptr, nullptr, qb, BS_, D_, D_);
        gemm_bf16<false, false, true><<<grid, 256, 0, stream>>>(xb, wkT, nullptr, nullptr, kb, BS_, D_, D_);
        gemm_bf16<false, false, true><<<grid, 256, 0, stream>>>(xb, wvT, nullptr, nullptr, vb, BS_, D_, D_);
    }

    // V transpose for attention B-fragments
    {
        dim3 grid(S_ / 32, DH_ / 32, B_ * H_);
        transpose_v<<<grid, 256, 0, stream>>>(vb, vt);
    }

    // flash attention -> ctx (bf16, heads merged)
    {
        dim3 grid(S_ / 16 / 4, B_ * H_);
        attn_kernel<<<grid, 128, 0, stream>>>(qb, kb, vt, vl, ctxb);
    }

    // output projection: [BS,D] x [D,D] -> f32
    {
        dim3 grid(D_ / 128, BS_ / 128);
        gemm_bf16<false, false, false><<<grid, 256, 0, stream>>>(ctxb, woT, nullptr, tf, nullptr, BS_, D_, D_);
    }

    // AddNorm 1: y = LN(x + attn_out) -> f32 + bf16
    addnorm_kernel<true><<<BS_, 128, 0, stream>>>(x, tf, g1, be1, y, yb);

    // FFN1: [BS,D] x [D,F] + b1, ReLU -> bf16 hidden
    {
        dim3 grid(F_ / 128, BS_ / 128);
        gemm_bf16<true, true, true><<<grid, 256, 0, stream>>>(yb, w1T, b1, nullptr, hb, BS_, F_, D_);
    }
    // FFN2: [BS,F] x [F,D] + b2 -> f32
    {
        dim3 grid(D_ / 128, BS_ / 128);
        gemm_bf16<true, false, false><<<grid, 256, 0, stream>>>(hb, w2T, b2, tf, nullptr, BS_, D_, F_);
    }

    // AddNorm 2 -> final output (f32)
    addnorm_kernel<false><<<BS_, 128, 0, stream>>>(y, tf, g2, be2, out, nullptr);

    (void)in_sizes; (void)n_in; (void)out_size; (void)ws_size;
}
